// ResidualGraphConv_19164144075227
// MI455X (gfx1250) — compile-verified, matrix operand order checked by reading, not run on hardware
//
#include <hip/hip_runtime.h>
#include <cstdint>
#include <cstddef>

// ---------------- types for WMMA ----------------
typedef __attribute__((ext_vector_type(16))) __bf16 v16bf;
typedef __attribute__((ext_vector_type(8)))  float  v8f;

union Frag16 {
    uint4  u4[2];
    v16bf  v;
};

#define BATCH 16
#define CIN   64
#define NN    2048
#define LDB   40            // padded LDS row stride (elements): 80B, 16B-aligned rows

__device__ inline unsigned pack2bf(float a, float b) {
    union { __bf16 h[2]; unsigned u; } p;
    p.h[0] = (__bf16)a;     // RNE f32->bf16; pairs fuse into v_cvt_pk_bf16_f32
    p.h[1] = (__bf16)b;
    return p.u;
}

// ---------------- fp32 -> bf16 convert (x -> xp0) ----------------
__global__ __launch_bounds__(256) void convert_x_kernel(const float* __restrict__ x,
                                                        __bf16* __restrict__ xb,
                                                        int total) {
    int i = blockIdx.x * 256 + threadIdx.x;
    if (i < total) xb[i] = (__bf16)x[i];
}

// ============================================================================
// Hop GEMM, fp32-adj path: Out[b] = A[b] (64xN bf16) x adj[b] (NxN f32->bf16)
// Double-buffered LDS staging (transposed, packed-dword stores).
// WRITE_T: additionally emit adjT[b][n][k] (bf16, transposed) for later hops.
// grid: (NN/128, BATCH), block 256 (8 waves)
// ============================================================================
template <bool WRITE_T>
__global__ __launch_bounds__(256) void gemm_hop_cvt_kernel(const __bf16* __restrict__ Abuf,
                                                           const float*  __restrict__ adj,
                                                           __bf16*       __restrict__ Out,
                                                           __bf16*       __restrict__ adjT) {
    __shared__ __bf16 ldsB[2][128 * LDB];   // adj chunk, transposed: [n][k]

    const int b    = blockIdx.y;
    const int nblk = blockIdx.x * 128;
    const int tid  = threadIdx.x;
    const int wave = tid >> 5;
    const int lane = tid & 31;
    const int half = lane >> 4;
    const int r    = lane & 15;

    const int mbase  = (wave & 3) * 16;     // 4 M-waves cover M=0..63
    const int nwbase = (wave >> 2) * 64;    // 2 N-waves cover 128 cols

    const __bf16* Ab   = Abuf + (size_t)b * CIN * NN;
    const float*  Adjb = adj  + (size_t)b * NN * NN;
    __bf16*       Tb   = WRITE_T ? (adjT + (size_t)b * NN * NN) : nullptr;

    // staging mapping: thread owns column sn, contiguous k-strip [skh, skh+16)
    const int sn  = tid & 127;
    const int skh = (tid >> 7) * 16;

    v8f c[4];
    c[0] = v8f{}; c[1] = v8f{}; c[2] = v8f{}; c[3] = v8f{};

    auto stage = [&](int k0, int buf) {
        unsigned pk[8];
        #pragma unroll
        for (int it = 0; it < 8; ++it) {
            const int kk = skh + it * 2;
            float a0 = Adjb[(size_t)(k0 + kk)     * NN + nblk + sn];
            float a1 = Adjb[(size_t)(k0 + kk + 1) * NN + nblk + sn];
            pk[it] = pack2bf(a0, a1);
            *(unsigned*)&ldsB[buf][sn * LDB + kk] = pk[it];
        }
        if (WRITE_T) {
            __bf16* dst = Tb + (size_t)(nblk + sn) * NN + k0 + skh;
            *(uint4*)(dst)     = make_uint4(pk[0], pk[1], pk[2], pk[3]);
            *(uint4*)(dst + 8) = make_uint4(pk[4], pk[5], pk[6], pk[7]);
        }
    };

    stage(0, 0);
    int buf = 0;
    for (int k0 = 0; k0 < NN; k0 += 32) {
        __syncthreads();                       // prev compute done; prev stage visible
        if (k0 + 32 < NN) stage(k0 + 32, buf ^ 1);

        // issue ALL fragment loads for this chunk, then compute
        Frag16 fa, fb[4];
        const __bf16* arow = Ab + (size_t)(mbase + r) * NN + k0 + half * 8;
        fa.u4[0] = *(const uint4*)(arow);
        fa.u4[1] = *(const uint4*)(arow + 16);
        #pragma unroll
        for (int j = 0; j < 4; ++j) {
            const __bf16* brow = &ldsB[buf][(nwbase + j * 16 + r) * LDB + half * 8];
            fb[j].u4[0] = *(const uint4*)(brow);
            fb[j].u4[1] = *(const uint4*)(brow + 16);
        }
        __builtin_amdgcn_sched_barrier(0);     // keep loads above the WMMAs

        #pragma unroll
        for (int j = 0; j < 4; ++j) {
            c[j] = __builtin_amdgcn_wmma_f32_16x16x32_bf16(
                false, fa.v, false, fb[j].v, (short)0, c[j], false, false);
        }
        buf ^= 1;
    }

    __bf16* Ob = Out + (size_t)b * CIN * NN;
    #pragma unroll
    for (int j = 0; j < 4; ++j) {
        const int ncol = nblk + nwbase + j * 16 + r;
        #pragma unroll
        for (int v = 0; v < 8; ++v) {
            const int mrow = mbase + v + 8 * half;
            Ob[(size_t)mrow * NN + ncol] = (__bf16)c[j][v];
        }
    }
}

// ============================================================================
// Hop GEMM, bf16 transposed-adj path: no LDS, no barriers.
// Register double-buffered fragments, pipelined one K-chunk ahead.
// grid: (NN/128, BATCH), 256 thr (8 waves)
// ============================================================================
__global__ __launch_bounds__(256) void gemm_hop_T_kernel(const __bf16* __restrict__ Abuf,
                                                         const __bf16* __restrict__ adjT,
                                                         __bf16*       __restrict__ Out) {
    const int b    = blockIdx.y;
    const int nblk = blockIdx.x * 128;
    const int tid  = threadIdx.x;
    const int wave = tid >> 5;
    const int lane = tid & 31;
    const int half = lane >> 4;
    const int r    = lane & 15;

    const int mbase  = (wave & 3) * 16;
    const int nwbase = (wave >> 2) * 64;

    const __bf16* Ab = Abuf + (size_t)b * CIN * NN;
    const __bf16* Tb = adjT + (size_t)b * NN * NN;

    const __bf16* arowBase = Ab + (size_t)(mbase + r) * NN + half * 8;
    const __bf16* browBase[4];
    #pragma unroll
    for (int j = 0; j < 4; ++j)
        browBase[j] = Tb + (size_t)(nblk + nwbase + j * 16 + r) * NN + half * 8;

    v8f c[4];
    c[0] = v8f{}; c[1] = v8f{}; c[2] = v8f{}; c[3] = v8f{};

    Frag16 fa[2];
    Frag16 fb[2][4];

    auto ld = [&](int k0, int s) {
        fa[s].u4[0] = *(const uint4*)(arowBase + k0);
        fa[s].u4[1] = *(const uint4*)(arowBase + k0 + 16);
        #pragma unroll
        for (int j = 0; j < 4; ++j) {
            fb[s][j].u4[0] = *(const uint4*)(browBase[j] + k0);
            fb[s][j].u4[1] = *(const uint4*)(browBase[j] + k0 + 16);
        }
    };

    ld(0, 0);
    #pragma unroll 2
    for (int k0 = 0; k0 < NN; k0 += 32) {
        const int s = (k0 >> 5) & 1;
        if (k0 + 32 < NN) ld(k0 + 32, s ^ 1);   // prefetch next chunk's fragments
        __builtin_amdgcn_sched_barrier(0);      // keep prefetch loads above WMMAs

        #pragma unroll
        for (int j = 0; j < 4; ++j) {
            c[j] = __builtin_amdgcn_wmma_f32_16x16x32_bf16(
                false, fa[s].v, false, fb[s][j].v, (short)0, c[j], false, false);
        }
    }

    __bf16* Ob = Out + (size_t)b * CIN * NN;
    #pragma unroll
    for (int j = 0; j < 4; ++j) {
        const int ncol = nblk + nwbase + j * 16 + r;
        #pragma unroll
        for (int v = 0; v < 8; ++v) {
            const int mrow = mbase + v + 8 * half;
            Ob[(size_t)mrow * NN + ncol] = (__bf16)c[j][v];
        }
    }
}

// ---------------- fused W1/W2 + bias + relu epilogue ----------------
__global__ __launch_bounds__(128) void combine_kernel(const float*  __restrict__ x,
                                                      const __bf16* __restrict__ xp1,
                                                      const __bf16* __restrict__ xp2,
                                                      const __bf16* __restrict__ xp3,
                                                      const float*  __restrict__ W1,
                                                      const float*  __restrict__ b1,
                                                      const float*  __restrict__ W2,
                                                      const float*  __restrict__ b2,
                                                      float*        __restrict__ out) {
    const int b = blockIdx.y;
    const int n = blockIdx.x * 128 + threadIdx.x;

    float acc[64];
    #pragma unroll
    for (int o = 0; o < 32; ++o) { acc[o] = b1[o]; acc[o + 32] = b2[o]; }

    const float*  xb = x   + (size_t)b * CIN * NN;
    const __bf16* p1 = xp1 + (size_t)b * CIN * NN;
    const __bf16* p2 = xp2 + (size_t)b * CIN * NN;
    const __bf16* p3 = xp3 + (size_t)b * CIN * NN;

    #pragma unroll 1
    for (int ch = 0; ch < CIN; ++ch) {
        const float v0 = xb[(size_t)ch * NN + n];
        const float v1 = (float)p1[(size_t)ch * NN + n];
        const float v2 = (float)p2[(size_t)ch * NN + n];
        const float v3 = (float)p3[(size_t)ch * NN + n];
        #pragma unroll
        for (int o = 0; o < 32; ++o) {
            acc[o]      += W1[(o      ) * CIN + ch] * v0
                         + W1[(32 + o) * CIN + ch] * v1
                         + W1[(64 + o) * CIN + ch] * v2
                         + W1[(96 + o) * CIN + ch] * v3;
            acc[o + 32] += W2[(o      ) * CIN + ch] * v0
                         + W2[(32 + o) * CIN + ch] * v1
                         + W2[(64 + o) * CIN + ch] * v2
                         + W2[(96 + o) * CIN + ch] * v3;
        }
    }

    float* ob = out + (size_t)b * 64 * NN;
    #pragma unroll
    for (int o = 0; o < 32; ++o) {
        ob[(size_t)o * NN + n]        = acc[o];
        ob[(size_t)(o + 32) * NN + n] = fmaxf(acc[o + 32], 0.0f);
    }
}

// ---------------- launcher ----------------
extern "C" void kernel_launch(void* const* d_in, const int* in_sizes, int n_in,
                              void* d_out, int out_size, void* d_ws, size_t ws_size,
                              hipStream_t stream) {
    const float* adj = (const float*)d_in[0];   // [16, 2048, 2048]
    const float* x   = (const float*)d_in[1];   // [16, 64, 2048]
    const float* W1  = (const float*)d_in[2];   // [4, 32, 64]
    const float* b1  = (const float*)d_in[3];   // [32]
    const float* W2  = (const float*)d_in[4];   // [4, 32, 64]
    const float* b2  = (const float*)d_in[5];   // [32]
    float* out = (float*)d_out;                 // [16, 64, 2048]

    const size_t hopElems  = (size_t)BATCH * CIN * NN;      // 2,097,152
    const size_t hopBytes  = hopElems * sizeof(__bf16);     // 4 MB
    const size_t adjTBytes = (size_t)BATCH * NN * NN * 2;   // 128 MB

    __bf16* xpb0 = (__bf16*)d_ws;
    __bf16* xpb1 = xpb0 + hopElems;
    __bf16* xpb2 = xpb1 + hopElems;
    __bf16* xpb3 = xpb2 + hopElems;
    __bf16* adjT = (__bf16*)((char*)d_ws + 4 * hopBytes);

    const bool bigws = ws_size >= 4 * hopBytes + adjTBytes; // deterministic per run

    {
        int total = (int)hopElems;
        convert_x_kernel<<<(total + 255) / 256, 256, 0, stream>>>(x, xpb0, total);
    }

    dim3 gg(NN / 128, BATCH);
    if (bigws) {
        // hop1 converts adj -> bf16 transposed copy while computing; hops 2,3 are LDS-free
        gemm_hop_cvt_kernel<true><<<gg, 256, 0, stream>>>(xpb0, adj, xpb1, adjT);
        gemm_hop_T_kernel<<<gg, 256, 0, stream>>>(xpb1, adjT, xpb2);
        gemm_hop_T_kernel<<<gg, 256, 0, stream>>>(xpb2, adjT, xpb3);
    } else {
        gemm_hop_cvt_kernel<false><<<gg, 256, 0, stream>>>(xpb0, adj, xpb1, nullptr);
        gemm_hop_cvt_kernel<false><<<gg, 256, 0, stream>>>(xpb1, adj, xpb2, nullptr);
        gemm_hop_cvt_kernel<false><<<gg, 256, 0, stream>>>(xpb2, adj, xpb3, nullptr);
    }

    dim3 gc(NN / 128, BATCH);
    combine_kernel<<<gc, 128, 0, stream>>>(x, xpb1, xpb2, xpb3, W1, b1, W2, b2, out);
}
